// SelfAttention_19920058319405
// MI455X (gfx1250) — matrix operand order repaired and checked by reading, not compiled, not run on hardware
//
#include <hip/hip_runtime.h>
#include <hip/hip_bf16.h>
#include <stdint.h>

// ---------------------------------------------------------------------------
// Problem constants (reference: B=4, T=2048, C=1024, H=16, D=64)
// ---------------------------------------------------------------------------
#define BB 4
#define TT 2048
#define CC 1024
#define HH 16
#define DD 64
#define MM (BB * TT)          // 8192 rows for the projection GEMMs
#define KDIM CC               // 1024
#define NDIM CC               // 1024

typedef __attribute__((ext_vector_type(16))) __bf16 v16bf;
typedef __attribute__((ext_vector_type(8)))  __bf16 v8bf;
typedef __attribute__((ext_vector_type(8)))  float  v8f;

__device__ __forceinline__ v8f vzero8() {
    v8f z = {0.f, 0.f, 0.f, 0.f, 0.f, 0.f, 0.f, 0.f};
    return z;
}

// A-matrix fragment (16x32 bf16, M x K): lane half 0 holds K=0..7 & 16..23,
// lane half 1 holds K=8..15 & 24..31 (CDNA5 ISA 7.12.2 16-bit A layout).
__device__ __forceinline__ v16bf ld_fragA(const __bf16* rowp, int lhalf) {
    union { v16bf v; v8bf h[2]; } u;
    u.h[0] = *(const v8bf*)(rowp + lhalf * 8);
    u.h[1] = *(const v8bf*)(rowp + 16 + lhalf * 8);
    return u.v;
}

// B-matrix fragment (32x16 bf16, K x N) loaded from an [N][K]-stored tile:
// lane n = lane%16, half 0 holds K=0..7 & 16..23, half 1 holds K=8..15 & 24..31.
__device__ __forceinline__ v16bf ld_fragB(const __bf16* rowp, int lhalf) {
    union { v16bf v; v8bf h[2]; } u;
    u.h[0] = *(const v8bf*)(rowp + lhalf * 8);
    u.h[1] = *(const v8bf*)(rowp + 16 + lhalf * 8);
    return u.v;
}

__device__ __forceinline__ v8f wmma_bf16(v16bf a, v16bf b, v8f c) {
    // (neg_a, A, neg_b, B, c_mod, C, reuse_a, reuse_b)
    return __builtin_amdgcn_wmma_f32_16x16x32_bf16(false, a, false, b,
                                                   (short)0, c, false, false);
}

// ---------------------------------------------------------------------------
// CDNA5 async direct-to-LDS copy (GLOBAL_LOAD_ASYNC_TO_LDS_B128, ASYNCcnt).
// GV mode: per-lane 64-bit global VA in a VGPR pair, per-lane LDS byte offset
// in a VGPR (low 32 bits of the generic shared-space address).
// ---------------------------------------------------------------------------
__device__ __forceinline__ void async_ld_b128(const void* gsrc, void* ldst) {
    unsigned loff = (unsigned)(uintptr_t)ldst;   // LDS aperture: addr[31:0] = LDS offset
    asm volatile("global_load_async_to_lds_b128 %0, %1, off"
                 :: "v"(loff), "v"(gsrc)
                 : "memory");
}
__device__ __forceinline__ void wait_async_le4() {
    asm volatile("s_wait_asynccnt 0x4" ::: "memory");
}
__device__ __forceinline__ void wait_async_0() {
    asm volatile("s_wait_asynccnt 0x0" ::: "memory");
}

// ---------------------------------------------------------------------------
// Kernel 0: f32 -> bf16 bulk convert (8 elements / thread, b128 stores)
// ---------------------------------------------------------------------------
__global__ void __launch_bounds__(256)
cvt_f32_bf16_kernel(const float* __restrict__ src, __bf16* __restrict__ dst, int n) {
    int i = (blockIdx.x * 256 + threadIdx.x) * 8;
    if (i >= n) return;
    float4 f0 = *(const float4*)(src + i);
    float4 f1 = *(const float4*)(src + i + 4);
    v8bf r;
    r[0] = (__bf16)f0.x; r[1] = (__bf16)f0.y; r[2] = (__bf16)f0.z; r[3] = (__bf16)f0.w;
    r[4] = (__bf16)f1.x; r[5] = (__bf16)f1.y; r[6] = (__bf16)f1.z; r[7] = (__bf16)f1.w;
    *(v8bf*)(dst + i) = r;
}

// ---------------------------------------------------------------------------
// Kernel 1: tiled bf16 GEMM  Y[m,n] = sum_k A[m,k] * W[n,k] + bias[n]
// BM=128, BN=128, BK=32; 8 waves, each computes a 32x64 sub-tile (2x4 WMMA).
// Tiles stream HBM -> LDS via async DMA, double-buffered.
// mode 0: f32 out, row-major [M,N]            (output projection)
// mode 1: bf16 out, [B,H,T,D], * 1/sqrt(D)    (Q, pre-scaled)
// mode 2: bf16 out, [B,H,T,D]                 (K)
// mode 3: bf16 out, [B,H,D,T]                 (V, pre-transposed for flash)
// ---------------------------------------------------------------------------
#define GBM 128
#define GBN 128
#define GBK 32
#define GBKP 40   // padded LDS stride (bf16 elems): 80B, 16B-aligned, bank-skewed

__global__ void __launch_bounds__(256)
gemm_bf16_kernel(const __bf16* __restrict__ A, const __bf16* __restrict__ W,
                 const float* __restrict__ bias, void* __restrict__ out, int mode) {
    __shared__ __align__(16) __bf16 As[2][GBM][GBKP];
    __shared__ __align__(16) __bf16 Ws[2][GBN][GBKP];

    const int tid   = threadIdx.x;
    const int lane  = tid & 31;
    const int wid   = tid >> 5;
    const int lrow  = lane & 15;
    const int lhalf = lane >> 4;
    const int wm    = wid & 3;   // 4 wave-rows of 32
    const int wn    = wid >> 2;  // 2 wave-cols of 64
    const int m0    = blockIdx.x * GBM;
    const int n0    = blockIdx.y * GBN;

    // cooperative tile staging: 256 threads x 4 async b128 per K-tile
    const int lr = tid >> 2;            // 0..63
    const int lc = (tid & 3) * 8;       // 0,8,16,24
    const __bf16* Ab = A + (size_t)(m0 + lr) * KDIM + lc;
    const __bf16* Wb = W + (size_t)(n0 + lr) * KDIM + lc;

    v8f acc[2][4];
#pragma unroll
    for (int fm = 0; fm < 2; ++fm)
#pragma unroll
        for (int fn = 0; fn < 4; ++fn) acc[fm][fn] = vzero8();

    // prologue: launch async DMA for K-tile 0 into buffer 0
    async_ld_b128(Ab,                        &As[0][lr][lc]);
    async_ld_b128(Ab + (size_t)64 * KDIM,    &As[0][lr + 64][lc]);
    async_ld_b128(Wb,                        &Ws[0][lr][lc]);
    async_ld_b128(Wb + (size_t)64 * KDIM,    &Ws[0][lr + 64][lc]);

    const int nk = KDIM / GBK;  // 32
    for (int kt = 0; kt < nk; ++kt) {
        const int cur = kt & 1;
        const bool havenext = (kt + 1) < nk;
        if (havenext) {  // stream next K-tile into the other buffer while we compute
            const __bf16* An = Ab + (kt + 1) * GBK;
            const __bf16* Wn = Wb + (kt + 1) * GBK;
            async_ld_b128(An,                     &As[cur ^ 1][lr][lc]);
            async_ld_b128(An + (size_t)64 * KDIM, &As[cur ^ 1][lr + 64][lc]);
            async_ld_b128(Wn,                     &Ws[cur ^ 1][lr][lc]);
            async_ld_b128(Wn + (size_t)64 * KDIM, &Ws[cur ^ 1][lr + 64][lc]);
            wait_async_le4();   // cur's 4 DMAs done (in-order), next 4 still in flight
        } else {
            wait_async_0();
        }
        __syncthreads();

        v16bf af[2], wf[4];
#pragma unroll
        for (int fm = 0; fm < 2; ++fm)
            af[fm] = ld_fragA(&As[cur][wm * 32 + fm * 16 + lrow][0], lhalf);
#pragma unroll
        for (int fn = 0; fn < 4; ++fn)
            wf[fn] = ld_fragB(&Ws[cur][wn * 64 + fn * 16 + lrow][0], lhalf);
#pragma unroll
        for (int fm = 0; fm < 2; ++fm)
#pragma unroll
            for (int fn = 0; fn < 4; ++fn)
                acc[fm][fn] = wmma_bf16(af[fm], wf[fn], acc[fm][fn]);

        __syncthreads();   // all waves done with cur before it is refilled
    }

    // epilogue: C/D layout -> lane l, vgpr i maps to (row = i + 8*(l>=16), col = l%16)
#pragma unroll
    for (int fm = 0; fm < 2; ++fm) {
#pragma unroll
        for (int fn = 0; fn < 4; ++fn) {
            const int rbase = m0 + wm * 32 + fm * 16 + lhalf * 8;
            const int col   = n0 + wn * 64 + fn * 16 + lrow;
            const float bv  = bias[col];
#pragma unroll
            for (int i = 0; i < 8; ++i) {
                const int   row = rbase + i;
                const float val = acc[fm][fn][i] + bv;
                if (mode == 0) {
                    // final output: write-once, keep it out of L2 (TH=NT)
                    __builtin_nontemporal_store(val,
                        (float*)out + (size_t)row * NDIM + col);
                } else {
                    const int b = row >> 11;         // row / T
                    const int t = row & (TT - 1);
                    const int h = col >> 6;          // col / D
                    const int d = col & (DD - 1);
                    __bf16* o = (__bf16*)out;
                    if (mode == 1)
                        o[((size_t)(b * HH + h) * TT + t) * DD + d] = (__bf16)(val * 0.125f);
                    else if (mode == 2)
                        o[((size_t)(b * HH + h) * TT + t) * DD + d] = (__bf16)val;
                    else
                        o[((size_t)(b * HH + h) * DD + d) * TT + t] = (__bf16)val;
                }
            }
        }
    }
}

// ---------------------------------------------------------------------------
// Kernel 2: causal flash attention, one (b,h) per blockIdx.y.
// 128 queries / block (8 waves x 16 rows), 64-key tiles, online softmax.
// K/V tiles stream HBM -> LDS via async DMA, double-buffered.
// Q is pre-scaled by 1/sqrt(D); V arrives transposed [D][T].
// ---------------------------------------------------------------------------
#define FKP 72  // padded LDS stride for 64-wide tiles (144B, 16B-aligned)

__global__ void __launch_bounds__(256)
flash_attn_kernel(const __bf16* __restrict__ Qh, const __bf16* __restrict__ Kh,
                  const __bf16* __restrict__ Vt, __bf16* __restrict__ AO) {
    __shared__ __align__(16) __bf16 Ks[2][64][FKP];   // [buf][key][d]
    __shared__ __align__(16) __bf16 Vs[2][64][FKP];   // [buf][d][key] (pre-transposed)
    __shared__ __align__(16) __bf16 Ps[8][16][FKP];   // per-wave P staging

    const int tid   = threadIdx.x;
    const int lane  = tid & 31;
    const int wid   = tid >> 5;
    const int lrow  = lane & 15;
    const int lhalf = lane >> 4;
    const int qbase = blockIdx.x * 128;
    const int bh    = blockIdx.y;                     // b*H + h
    const size_t base = (size_t)bh * TT * DD;

    const int lr = tid >> 2;        // 0..63
    const int lc = (tid & 3) * 8;   // 0,8,16,24
    const __bf16* Kg = Kh + base;
    const __bf16* Vg = Vt + base;   // [D][T]

    // prologue: async DMA for key-tile 0 into buffer 0
    async_ld_b128(Kg + (size_t)lr * DD + lc,      &Ks[0][lr][lc]);
    async_ld_b128(Kg + (size_t)lr * DD + lc + 32, &Ks[0][lr][lc + 32]);
    async_ld_b128(Vg + (size_t)lr * TT + lc,      &Vs[0][lr][lc]);
    async_ld_b128(Vg + (size_t)lr * TT + lc + 32, &Vs[0][lr][lc + 32]);

    // Q fragments live in registers for the whole kernel (A layout, direct load)
    const __bf16* qp = Qh + base + (size_t)(qbase + wid * 16 + lrow) * DD;
    v16bf qf[2];
    qf[0] = ld_fragA(qp, lhalf);
    qf[1] = ld_fragA(qp + 32, lhalf);

    float mi[8], li[8];
    v8f o[4];
#pragma unroll
    for (int i = 0; i < 8; ++i) { mi[i] = -3.0e38f; li[i] = 0.f; }
#pragma unroll
    for (int f = 0; f < 4; ++f) o[f] = vzero8();

    const int nkt = (qbase + 128) / 64;   // causal: only tiles up to the diagonal
    for (int kt = 0; kt < nkt; ++kt) {
        const int cur = kt & 1;
        const int kb  = kt * 64;
        const bool havenext = (kt + 1) < nkt;
        if (havenext) {
            const int nb = kb + 64;
            async_ld_b128(Kg + (size_t)(nb + lr) * DD + lc,      &Ks[cur ^ 1][lr][lc]);
            async_ld_b128(Kg + (size_t)(nb + lr) * DD + lc + 32, &Ks[cur ^ 1][lr][lc + 32]);
            async_ld_b128(Vg + (size_t)lr * TT + nb + lc,        &Vs[cur ^ 1][lr][lc]);
            async_ld_b128(Vg + (size_t)lr * TT + nb + lc + 32,   &Vs[cur ^ 1][lr][lc + 32]);
            wait_async_le4();
        } else {
            wait_async_0();
        }
        __syncthreads();

        // S = Q K^T : 16q x 64k per wave, 8 WMMA
        v8f s[4];
#pragma unroll
        for (int kf = 0; kf < 4; ++kf) s[kf] = vzero8();
#pragma unroll
        for (int kk = 0; kk < 2; ++kk) {
#pragma unroll
            for (int kf = 0; kf < 4; ++kf) {
                v16bf kfrag = ld_fragB(&Ks[cur][kf * 16 + lrow][kk * 32], lhalf);
                s[kf] = wmma_bf16(qf[kk], kfrag, s[kf]);
            }
        }

        // causal mask (scores layout: vgpr i -> row = base + i + 8*lhalf, col = lane%16)
        const int rbase = qbase + wid * 16 + lhalf * 8;
#pragma unroll
        for (int kf = 0; kf < 4; ++kf) {
            const int col = kb + kf * 16 + lrow;
#pragma unroll
            for (int i = 0; i < 8; ++i)
                if (col > rbase + i) s[kf][i] = -3.0e38f;
        }

        // online softmax: row reductions across the 16-lane half via shfl_xor
        float alpha[8];
#pragma unroll
        for (int i = 0; i < 8; ++i) {
            float mx = fmaxf(fmaxf(s[0][i], s[1][i]), fmaxf(s[2][i], s[3][i]));
            mx = fmaxf(mx, __shfl_xor(mx, 1, 32));
            mx = fmaxf(mx, __shfl_xor(mx, 2, 32));
            mx = fmaxf(mx, __shfl_xor(mx, 4, 32));
            mx = fmaxf(mx, __shfl_xor(mx, 8, 32));
            const float mnew = fmaxf(mi[i], mx);
            alpha[i] = __expf(mi[i] - mnew);
            mi[i] = mnew;
        }
#pragma unroll
        for (int kf = 0; kf < 4; ++kf)
#pragma unroll
            for (int i = 0; i < 8; ++i)
                s[kf][i] = __expf(s[kf][i] - mi[i]);
#pragma unroll
        for (int i = 0; i < 8; ++i) {
            float rs = s[0][i] + s[1][i] + s[2][i] + s[3][i];
            rs += __shfl_xor(rs, 1, 32);
            rs += __shfl_xor(rs, 2, 32);
            rs += __shfl_xor(rs, 4, 32);
            rs += __shfl_xor(rs, 8, 32);
            li[i] = li[i] * alpha[i] + rs;
        }
#pragma unroll
        for (int f = 0; f < 4; ++f)
#pragma unroll
            for (int i = 0; i < 8; ++i) o[f][i] *= alpha[i];

        // stage P (bf16) into per-wave LDS; same-wave DS ops are in-order
#pragma unroll
        for (int kf = 0; kf < 4; ++kf)
#pragma unroll
            for (int i = 0; i < 8; ++i)
                Ps[wid][i + lhalf * 8][kf * 16 + lrow] = (__bf16)s[kf][i];
        asm volatile("s_wait_dscnt 0x0" ::: "memory");

        // O += P V : 8 WMMA (A = P frags, B = V^T tile already [d][key])
#pragma unroll
        for (int kk = 0; kk < 2; ++kk) {
            v16bf pa = ld_fragA(&Ps[wid][lrow][kk * 32], lhalf);
#pragma unroll
            for (int f = 0; f < 4; ++f) {
                v16bf vb = ld_fragB(&Vs[cur][f * 16 + lrow][kk * 32], lhalf);
                o[f] = wmma_bf16(pa, vb, o[f]);
            }
        }
        __syncthreads();   // all waves done with cur before it is refilled
    }

    // epilogue: O /= l, write attn output as [B,T,C] bf16 (ready for projection)
    const int b = bh >> 4, h = bh & (HH - 1);
#pragma unroll
    for (int i = 0; i < 8; ++i) li[i] = 1.f / li[i];
#pragma unroll
    for (int f = 0; f < 4; ++f) {
#pragma unroll
        for (int i = 0; i < 8; ++i) {
            const int q = qbase + wid * 16 + i + lhalf * 8;
            AO[((size_t)b * TT + q) * CC + h * DD + f * 16 + lrow] =
                (__bf16)(o[f][i] * li[i]);
        }
    }
}

// ---------------------------------------------------------------------------
// Host-side launcher
// ---------------------------------------------------------------------------
extern "C" void kernel_launch(void* const* d_in, const int* in_sizes, int n_in,
                              void* d_out, int out_size, void* d_ws, size_t ws_size,
                              hipStream_t stream) {
    const float* x  = (const float*)d_in[0];
    const float* Wk = (const float*)d_in[1];
    const float* bk = (const float*)d_in[2];
    const float* Wq = (const float*)d_in[3];
    const float* bq = (const float*)d_in[4];
    const float* Wv = (const float*)d_in[5];
    const float* bv = (const float*)d_in[6];
    const float* Wp = (const float*)d_in[7];
    const float* bp = (const float*)d_in[8];

    char* ws = (char*)d_ws;
    size_t off = 0;
    auto alloc = [&](size_t bytes) -> void* {
        void* p = ws + off;
        off += (bytes + 255) & ~(size_t)255;
        return p;
    };
    __bf16* xb  = (__bf16*)alloc((size_t)MM * CC * 2);
    __bf16* Wkb = (__bf16*)alloc((size_t)CC * CC * 2);
    __bf16* Wqb = (__bf16*)alloc((size_t)CC * CC * 2);
    __bf16* Wvb = (__bf16*)alloc((size_t)CC * CC * 2);
    __bf16* Wpb = (__bf16*)alloc((size_t)CC * CC * 2);
    __bf16* Qh  = (__bf16*)alloc((size_t)BB * HH * TT * DD * 2);  // [B,H,T,D]
    __bf16* Kh  = (__bf16*)alloc((size_t)BB * HH * TT * DD * 2);  // [B,H,T,D]
    __bf16* Vt  = (__bf16*)alloc((size_t)BB * HH * DD * TT * 2);  // [B,H,D,T]
    __bf16* AO  = (__bf16*)alloc((size_t)MM * CC * 2);            // [B,T,C]

    auto cvt = [&](const float* s, __bf16* d, int n) {
        cvt_f32_bf16_kernel<<<dim3((n / 8 + 255) / 256), dim3(256), 0, stream>>>(s, d, n);
    };
    cvt(x,  xb,  MM * CC);
    cvt(Wk, Wkb, CC * CC);
    cvt(Wq, Wqb, CC * CC);
    cvt(Wv, Wvb, CC * CC);
    cvt(Wp, Wpb, CC * CC);

    const dim3 ggrid(MM / GBM, NDIM / GBN);   // (64, 8)
    gemm_bf16_kernel<<<ggrid, dim3(256), 0, stream>>>(xb, Wkb, bk, (void*)Kh, 2);
    gemm_bf16_kernel<<<ggrid, dim3(256), 0, stream>>>(xb, Wqb, bq, (void*)Qh, 1);
    gemm_bf16_kernel<<<ggrid, dim3(256), 0, stream>>>(xb, Wvb, bv, (void*)Vt, 3);

    flash_attn_kernel<<<dim3(TT / 128, BB * HH), dim3(256), 0, stream>>>(Qh, Kh, Vt, AO);

    gemm_bf16_kernel<<<ggrid, dim3(256), 0, stream>>>(AO, Wpb, bp, d_out, 0);
}
